// MinimalLinearAttention_48601849921698
// MI455X (gfx1250) — matrix operand order repaired and checked by reading, not compiled, not run on hardware
//
#include <hip/hip_runtime.h>
#include <hip/hip_bf16.h>

typedef __attribute__((ext_vector_type(8)))  __bf16         v8bf;
typedef __attribute__((ext_vector_type(16))) __bf16         v16bf;
typedef __attribute__((ext_vector_type(8)))  float          v8f;
typedef __attribute__((ext_vector_type(8)))  unsigned short v8us;
typedef __attribute__((ext_vector_type(4)))  unsigned int   ui4;
typedef __attribute__((ext_vector_type(8)))  int            i8v;
typedef __attribute__((ext_vector_type(4)))  int            i4v;

#define BB 4
#define SS 4096
#define DD 1024
#define HH 16
#define HD 64

__device__ __forceinline__ unsigned short f2bf(float f) {
  unsigned int u = __builtin_bit_cast(unsigned int, f);
  u += 0x7fffu + ((u >> 16) & 1u);          // round-to-nearest-even
  return (unsigned short)(u >> 16);
}

__device__ __forceinline__ v16bf cat16(v8bf lo, v8bf hi) {
  return __builtin_shufflevector(lo, hi, 0, 1, 2, 3, 4, 5, 6, 7,
                                 8, 9, 10, 11, 12, 13, 14, 15);
}

// A-matrix fragment (16x32 bf16, M x K). lane<16 -> M=lane, K = 0..15;
// lane>=16 -> M=lane-16, K = 16..31 (contiguous 16 halfs, matches frag_b).
__device__ __forceinline__ v16bf frag_a(const unsigned short* p0, int stride, int lane) {
  const int l = lane & 15, hi = lane >> 4;
  const unsigned short* p = p0 + (size_t)l * stride + hi * 16;
  v8bf x0 = *(const v8bf*)p;
  v8bf x1 = *(const v8bf*)(p + 8);
  return cat16(x0, x1);
}

// B-matrix fragment (32x16 bf16, K x N). Lane n (0..15) holds column n,
// K=0..15; lane n+16 holds column n, K=16..31 (contiguous 16 halfs).
__device__ __forceinline__ v16bf frag_b(const unsigned short* p0, int stride, int lane) {
  const int l = lane & 15, hi = lane >> 4;
  const unsigned short* p = p0 + (size_t)l * stride + hi * 16;
  v8bf x0 = *(const v8bf*)p;
  v8bf x1 = *(const v8bf*)(p + 8);
  return cat16(x0, x1);
}

__device__ __forceinline__ v8f wmma_bf16(v16bf a, v16bf b, v8f c) {
  return __builtin_amdgcn_wmma_f32_16x16x32_bf16(false, a, false, b,
                                                 (short)0, c, false, false);
}

__device__ __forceinline__ v8f v8f_zero() {
  v8f z = {0.f, 0.f, 0.f, 0.f, 0.f, 0.f, 0.f, 0.f};
  return z;
}

// --- CDNA5 async global->LDS copy (GLOBAL_LOAD_ASYNC_TO_LDS_B128, ASYNCcnt).
__device__ __forceinline__ void async_ld_b128(unsigned int lds_addr,
                                              const unsigned short* g) {
  asm volatile("global_load_async_to_lds_b128 %0, %1, off"
               :: "v"(lds_addr), "v"((unsigned long long)(uintptr_t)g)
               : "memory");
}
__device__ __forceinline__ void wait_async0() {
  asm volatile("s_wait_asynccnt 0x0" ::: "memory");
}
__device__ __forceinline__ void wait_async4() {
  asm volatile("s_wait_asynccnt 0x4" ::: "memory");
}

// --- CDNA5 Tensor Data Mover: one 2-D bf16 tile -> LDS (TENSORcnt).
// D# per ISA 08 §8.3/8.4. pad_enable with pad_interval=3 (16 DWORDs = one
// 64B tile row) and pad_amount=3 (4 DWORDs = 16B) makes the TDM write LDS
// rows at an 80-byte stride == the padded layout frag_a/frag_b read.
// This toolchain declares the 6-arg builtin: (g0, g1, g2, g3, g_extra, cpol).
__device__ __forceinline__ void tdm_load_2d(unsigned int lds_addr,
                                            const unsigned short* gaddr,
                                            unsigned int tensor_d0,  // line len (elems)
                                            unsigned int tensor_d1,  // # lines
                                            unsigned int stride_d0,  // elems
                                            unsigned int tile_d0,    // elems
                                            unsigned int tile_d1) {  // lines
  const unsigned long long ga = (unsigned long long)(uintptr_t)gaddr;
  ui4 g0;
  g0[0] = 1u;                                          // count=1, user load
  g0[1] = lds_addr;                                    // lds_addr[31:0]
  g0[2] = (unsigned int)(ga & 0xffffffffu);            // global_addr[31:0]
  g0[3] = (unsigned int)((ga >> 32) & 0x01ffffffu)     // global_addr[56:32]
          | (2u << 30);                                // type = 2 ("image")
  i8v g1;
  g1[0] = (int)((1u << 16)        // data_size = 1 -> 2 bytes
                | (1u << 20)      // pad_enable
                | (3u << 22)      // pad_interval: 16 DWORDs (64B row)
                | (3u << 25));    // pad_amount:   4 DWORDs (16B)
  g1[1] = (int)((tensor_d0 & 0xffffu) << 16);                       // dim0 lo
  g1[2] = (int)((tensor_d0 >> 16) | ((tensor_d1 & 0xffffu) << 16)); // dim0 hi, dim1 lo
  g1[3] = (int)((tensor_d1 >> 16) | (tile_d0 << 16));               // dim1 hi, tile0
  g1[4] = (int)(tile_d1 & 0xffffu);                                 // tile1, tile2=0
  g1[5] = (int)stride_d0;                                           // dim0 stride lo
  g1[6] = 0;                                                        // stride hi
  g1[7] = 0;
  i4v z4 = {0, 0, 0, 0};
  i8v z8 = {0, 0, 0, 0, 0, 0, 0, 0};
  __builtin_amdgcn_tensor_load_to_lds(g0, g1, z4, z4, z8, 0);
}

// ---------------------------------------------------------------------------
// fp32 -> bf16 conversion (grid-stride)
// ---------------------------------------------------------------------------
__global__ void cvt_bf16_kernel(const float* __restrict__ in,
                                unsigned short* __restrict__ out, int n) {
  for (int i = blockIdx.x * blockDim.x + threadIdx.x; i < n;
       i += gridDim.x * blockDim.x)
    out[i] = f2bf(in[i]);
}

// ---------------------------------------------------------------------------
// C[m,n] = act( sum_k A[m,k] * W[n,k] + bias[n] ), A,W bf16, acc fp32.
// 256x128 tile / workgroup; 8 waves as 4(m)x2(n), each wave owns 64x64
// (16 WMMAs per K-step). Double-buffered LDS staging:
//   A tile (256x32): per-thread GLOBAL_LOAD_ASYNC_TO_LDS_B128 (ASYNCcnt)
//   B tile (128x32): one TENSOR_LOAD_TO_LDS descriptor from wave 0 (TENSORcnt)
// s_wait_asynccnt 4 / s_wait_tensorcnt 1 keep the next slab in flight.
// ---------------------------------------------------------------------------
template <bool RELU, bool OUTF32>
__global__ __launch_bounds__(256) void gemm_wmma_kernel(
    const unsigned short* __restrict__ A, const unsigned short* __restrict__ W,
    const float* __restrict__ bias, void* __restrict__ Cout,
    int M, int N, int K) {
  __shared__ unsigned short sA[2][256 * 40];   // rows padded to 40 halfs (80B)
  __shared__ unsigned short sB[2][128 * 40];

  const int tid  = threadIdx.x;
  const int lane = tid & 31;
  const int wave = tid >> 5;
  const int wm   = wave >> 1;   // 0..3 : 64 output rows each
  const int wn   = wave & 1;    // 0..1 : 64 output cols each
  const int m0   = blockIdx.y * 256;
  const int n0   = blockIdx.x * 128;

  v8f acc[4][4];
#pragma unroll
  for (int mi = 0; mi < 4; ++mi)
#pragma unroll
    for (int ni = 0; ni < 4; ++ni) acc[mi][ni] = v8f_zero();

  const unsigned short* ga = A + (size_t)(m0 + tid) * K;   // A: one row/thread
  const unsigned short* gw = W + (size_t)n0 * K;           // B tile base

  unsigned int aoff[2], bbase[2];
  aoff[0]  = (unsigned int)(uintptr_t)&sA[0][tid * 40];
  aoff[1]  = (unsigned int)(uintptr_t)&sA[1][tid * 40];
  bbase[0] = (unsigned int)(uintptr_t)&sB[0][0];
  bbase[1] = (unsigned int)(uintptr_t)&sB[1][0];

  auto issueA = [&](int buf, int k0) {       // 4 async b128 per thread
    async_ld_b128(aoff[buf],      ga + k0);
    async_ld_b128(aoff[buf] + 16, ga + k0 + 8);
    async_ld_b128(aoff[buf] + 32, ga + k0 + 16);
    async_ld_b128(aoff[buf] + 48, ga + k0 + 24);
  };
  auto issueB = [&](int buf, int k0) {       // one TDM descriptor (wave 0)
    tdm_load_2d(bbase[buf], gw + k0, (unsigned int)K, 128u,
                (unsigned int)K, 32u, 128u);
  };

  issueA(0, 0);
  if (tid < 32) issueB(0, 0);
  int cur = 0;
  for (int k0 = 0; k0 < K; k0 += 32) {
    if (k0 + 32 < K) {                       // software pipeline: next slab
      issueA(cur ^ 1, k0 + 32);
      if (tid < 32) issueB(cur ^ 1, k0 + 32);
      wait_async4();                         // this thread's current A done
      if (tid < 32) __builtin_amdgcn_s_wait_tensorcnt(1);  // current B done
    } else {
      wait_async0();
      if (tid < 32) __builtin_amdgcn_s_wait_tensorcnt(0);
    }
    __syncthreads();

    v16bf af[4], bfg[4];
#pragma unroll
    for (int mi = 0; mi < 4; ++mi)
      af[mi] = frag_a(&sA[cur][(wm * 64 + mi * 16) * 40], 40, lane);
#pragma unroll
    for (int ni = 0; ni < 4; ++ni)
      bfg[ni] = frag_b(&sB[cur][(wn * 64 + ni * 16) * 40], 40, lane);
#pragma unroll
    for (int mi = 0; mi < 4; ++mi)
#pragma unroll
      for (int ni = 0; ni < 4; ++ni)
        acc[mi][ni] = wmma_bf16(af[mi], bfg[ni], acc[mi][ni]);
    __syncthreads();             // all waves done reading before buf reuse
    cur ^= 1;
  }

  const int l16 = lane & 15;
  const int hi  = lane >> 4;
#pragma unroll
  for (int mi = 0; mi < 4; ++mi) {
    const int rbase = m0 + wm * 64 + mi * 16 + hi * 8;
#pragma unroll
    for (int ni = 0; ni < 4; ++ni) {
      const int col = n0 + wn * 64 + ni * 16 + l16;
      const float bv = bias[col];
#pragma unroll
      for (int j = 0; j < 8; ++j) {
        float v = acc[mi][ni][j] + bv;
        if (RELU) v = v > 0.f ? v : 0.f;
        const size_t idx = (size_t)(rbase + j) * N + col;
        if (OUTF32)
          ((float*)Cout)[idx] = v;
        else
          ((unsigned short*)Cout)[idx] = f2bf(v);
      }
    }
  }
}

// ---------------------------------------------------------------------------
// KV state: per (b,h), KV[d][e] = sum_s K[s,d] * V[s,e]  (64x64, red. S=4096)
// Stage K,V head slices transposed into LDS so WMMA A/B frags are contiguous.
// 8 waves: (sgrp = S-split of each chunk) x (mrow = 16-row output strip);
// cross-wave combine via ds_add_f32. Output stored transposed: KVt[e][d].
// ---------------------------------------------------------------------------
__global__ __launch_bounds__(256) void kv_state_kernel(
    const unsigned short* __restrict__ Km, const unsigned short* __restrict__ Vm,
    unsigned short* __restrict__ KVt) {
  __shared__ unsigned short sKt[64 * 136];   // [d][s] padded rows (272B)
  __shared__ unsigned short sVt[64 * 136];   // [e][s]
  __shared__ float sAcc[64 * 64];

  const int bh = blockIdx.x;
  const int b = bh >> 4, h = bh & 15;
  const int tid = threadIdx.x, lane = tid & 31, wave = tid >> 5;
  const int sgrp = wave >> 2, mrow = wave & 3;

  for (int i = tid; i < 64 * 64; i += 256) sAcc[i] = 0.f;

  v8f acc[4];
#pragma unroll
  for (int ni = 0; ni < 4; ++ni) acc[ni] = v8f_zero();

  const size_t base = (size_t)b * SS * DD + (size_t)h * HD;
  const int si = tid >> 1;          // 0..127 token within chunk
  const int d0 = (tid & 1) * 32;    // 0 or 32

  for (int c = 0; c < SS; c += 128) {
    const unsigned short* gk = Km + base + (size_t)(c + si) * DD + d0;
    const unsigned short* gv = Vm + base + (size_t)(c + si) * DD + d0;
#pragma unroll
    for (int g = 0; g < 4; ++g) {
      v8us kk = *(const v8us*)(gk + g * 8);
      v8us vv = *(const v8us*)(gv + g * 8);
#pragma unroll
      for (int e = 0; e < 8; ++e) {
        sKt[(d0 + g * 8 + e) * 136 + si] = kk[e];
        sVt[(d0 + g * 8 + e) * 136 + si] = vv[e];
      }
    }
    __syncthreads();
#pragma unroll
    for (int ks = 0; ks < 2; ++ks) {
      const int kk0 = (sgrp * 2 + ks) * 32;
      v16bf af = frag_a(&sKt[(mrow * 16) * 136 + kk0], 136, lane);
#pragma unroll
      for (int ni = 0; ni < 4; ++ni) {
        v16bf bfg = frag_b(&sVt[(ni * 16) * 136 + kk0], 136, lane);
        acc[ni] = wmma_bf16(af, bfg, acc[ni]);
      }
    }
    __syncthreads();
  }

  const int l16 = lane & 15, hi = lane >> 4;
#pragma unroll
  for (int ni = 0; ni < 4; ++ni) {
    const int n = ni * 16 + l16;
#pragma unroll
    for (int j = 0; j < 8; ++j) {
      const int m = mrow * 16 + j + hi * 8;
      atomicAdd(&sAcc[m * 64 + n], acc[ni][j]);   // ds_add_f32
    }
  }
  __syncthreads();

  unsigned short* outp = KVt + (size_t)bh * 64 * 64;
  for (int i = tid; i < 64 * 64; i += 256) {
    const int e = i >> 6, d = i & 63;
    outp[i] = f2bf(sAcc[d * 64 + e]);             // store transposed KVt[e][d]
  }
}

// ---------------------------------------------------------------------------
// Readout: AO[s, h*64+e] = sum_d Q[s,h*64+d] * KV[d][e]   (M=128/block, K=64)
// ---------------------------------------------------------------------------
__global__ __launch_bounds__(256) void readout_kernel(
    const unsigned short* __restrict__ Q, const unsigned short* __restrict__ KVt,
    unsigned short* __restrict__ AO) {
  const int bh = blockIdx.y;
  const int b = bh >> 4, h = bh & 15;
  const int tid = threadIdx.x, lane = tid & 31, wave = tid >> 5;
  const size_t row0 = (size_t)b * SS + (size_t)blockIdx.x * 128 + wave * 16;
  const unsigned short* qb = Q + row0 * DD + h * HD;
  const unsigned short* kv = KVt + (size_t)bh * 64 * 64;   // [e][d]
  unsigned short* ob = AO + row0 * DD + h * HD;

  v8f acc[4];
#pragma unroll
  for (int ni = 0; ni < 4; ++ni) acc[ni] = v8f_zero();

#pragma unroll
  for (int ks = 0; ks < 2; ++ks) {
    const int k0 = ks * 32;
    v16bf af = frag_a(qb + k0, DD, lane);
#pragma unroll
    for (int ni = 0; ni < 4; ++ni) {
      v16bf bfg = frag_b(kv + (ni * 16) * 64 + k0, 64, lane);
      acc[ni] = wmma_bf16(af, bfg, acc[ni]);
    }
  }

  const int l16 = lane & 15, hi = lane >> 4;
#pragma unroll
  for (int ni = 0; ni < 4; ++ni) {
    const int n = ni * 16 + l16;
#pragma unroll
    for (int j = 0; j < 8; ++j)
      ob[(size_t)(j + hi * 8) * DD + n] = f2bf(acc[ni][j]);
  }
}

// ---------------------------------------------------------------------------
extern "C" void kernel_launch(void* const* d_in, const int* in_sizes, int n_in,
                              void* d_out, int out_size, void* d_ws, size_t ws_size,
                              hipStream_t stream) {
  const float* x   = (const float*)d_in[0];
  const float* q_w = (const float*)d_in[1];
  const float* q_b = (const float*)d_in[2];
  const float* k_w = (const float*)d_in[3];
  const float* k_b = (const float*)d_in[4];
  const float* v_w = (const float*)d_in[5];
  const float* v_b = (const float*)d_in[6];
  const float* o_w = (const float*)d_in[7];
  const float* o_b = (const float*)d_in[8];

  const size_t MX = (size_t)BB * SS;   // 16384 token rows
  const size_t XE = MX * DD;           // elems in x / q / k / v / ao
  const size_t WE = (size_t)DD * DD;   // elems per weight matrix

  unsigned short* Xb  = (unsigned short*)d_ws;  // bf16 x  (reused for AO later)
  unsigned short* Qw  = Xb + XE;
  unsigned short* Kw  = Qw + WE;
  unsigned short* Vw  = Kw + WE;
  unsigned short* Ow  = Vw + WE;
  unsigned short* Qm  = Ow + WE;
  unsigned short* Km  = Qm + XE;
  unsigned short* Vm  = Km + XE;
  unsigned short* KVt = Vm + XE;                // B*H * 64*64
  unsigned short* AO  = Xb;                     // alias: x dead after QKV GEMMs

  dim3 blk(256);
  cvt_bf16_kernel<<<2048, blk, 0, stream>>>(x,   Xb, (int)XE);
  cvt_bf16_kernel<<<512,  blk, 0, stream>>>(q_w, Qw, (int)WE);
  cvt_bf16_kernel<<<512,  blk, 0, stream>>>(k_w, Kw, (int)WE);
  cvt_bf16_kernel<<<512,  blk, 0, stream>>>(v_w, Vw, (int)WE);
  cvt_bf16_kernel<<<512,  blk, 0, stream>>>(o_w, Ow, (int)WE);

  dim3 ggrid(DD / 128, (int)(MX / 256));        // (8, 64)
  gemm_wmma_kernel<true,  false><<<ggrid, blk, 0, stream>>>(Xb, Qw, q_b, Qm, (int)MX, DD, DD);
  gemm_wmma_kernel<true,  false><<<ggrid, blk, 0, stream>>>(Xb, Kw, k_b, Km, (int)MX, DD, DD);
  gemm_wmma_kernel<false, false><<<ggrid, blk, 0, stream>>>(Xb, Vw, v_b, Vm, (int)MX, DD, DD);

  kv_state_kernel<<<BB * HH, blk, 0, stream>>>(Km, Vm, KVt);

  dim3 rgrid(SS / 128, BB * HH);                // (32, 64)
  readout_kernel<<<rgrid, blk, 0, stream>>>(Qm, KVt, AO);

  gemm_wmma_kernel<false, true><<<ggrid, blk, 0, stream>>>(AO, Ow, o_b, d_out, (int)MX, DD, DD);
}
